// Attention_16621523436388
// MI455X (gfx1250) — compile-verified, matrix operand order checked by reading, not run on hardware
//
#include <hip/hip_runtime.h>
#include <cstdint>
#include <cstddef>

typedef __attribute__((ext_vector_type(16))) __bf16 v16bf;
typedef __attribute__((ext_vector_type(8)))  float  v8f;
typedef __attribute__((ext_vector_type(4)))  int    v4i;

#define NSEQ   16
#define QLEN   128
#define NHEAD  8
#define NGRP   4
#define DHEAD  128
#define PAGESZ 64
#define PPS    16

union Frag { v16bf v; uint32_t u[8]; };

__device__ __forceinline__ uint32_t bf16of(float f) {
  uint32_t u = __builtin_bit_cast(uint32_t, f);
  return (u + 0x7fffu + ((u >> 16) & 1u)) >> 16;   // round-to-nearest-even
}
__device__ __forceinline__ uint32_t pack2(float a, float b) {
  return bf16of(a) | (bf16of(b) << 16);
}

// ---- CDNA5 async LDS-DMA (ASYNCcnt path), guarded so compile never breaks ----
__device__ __forceinline__ void async_ld16(const float* g, float* l) {
#if __has_builtin(__builtin_amdgcn_global_load_async_to_lds_b128)
  __builtin_amdgcn_global_load_async_to_lds_b128(
      (v4i*)(void*)g, (v4i*)(void*)l, 0, 0);
#else
  *reinterpret_cast<float4*>(l) = *reinterpret_cast<const float4*>(g);
#endif
}
__device__ __forceinline__ void wait_async() {
#if __has_builtin(__builtin_amdgcn_s_wait_asynccnt)
  __builtin_amdgcn_s_wait_asynccnt(0);
#endif
}

__device__ __forceinline__ void issue_chunk_load(const float* __restrict__ kv,
                                                 int page, int h, int tid,
                                                 float* Ks, float* Vs) {
  const float* base = kv + (size_t)page * PAGESZ * (2 * NHEAD) * DHEAD;
#pragma unroll
  for (int i = 0; i < 8; ++i) {
    int slot = tid + (i << 8);                 // 2048 slots of 16B per matrix
    int n = slot >> 5, dq = (slot & 31) << 2;
    async_ld16(base + ((size_t)n * 16 + h) * 128 + dq,     Ks + n * 128 + dq);
    async_ld16(base + ((size_t)n * 16 + 8 + h) * 128 + dq, Vs + n * 128 + dq);
  }
}

__global__ __launch_bounds__(256, 2) void ragged_paged_attn_kernel(
    const float* __restrict__ q, const float* __restrict__ kv,
    const int* __restrict__ kv_lens, const int* __restrict__ page_idx,
    float* __restrict__ out)
{
  __shared__ float           Kstage[64 * 128];  // f32 async staging (K)  32KB
  __shared__ float           Vstage[64 * 128];  // f32 async staging (V)  32KB
  __shared__ uint32_t        Kl[64 * 64];       // bf16 pairs [key][d/2]  16KB
  __shared__ unsigned short  Vt[128 * 64];      // bf16 transposed [d][k] 16KB
  __shared__ unsigned short  Pl[8 * 16 * 64];   // per-wave P [w][row][k] 16KB

  const int tid  = threadIdx.x;
  const int lane = tid & 31, wave = tid >> 5;
  const int half = lane >> 4, ln = lane & 15;
  const int bx = blockIdx.x;
  const int s = bx >> 5, hg = bx & 31, h = hg >> 2, g = hg & 3;
  const int m0 = wave * 16;
  const int kvlen = kv_lens[s];
  const float sm = 0.08838834764831845f;        // 1/sqrt(128)
  const int nchunks = (kvlen + 63) >> 6;

  // kick off async stage of chunk 0 (overlaps with Q preload below)
  issue_chunk_load(kv, page_idx[s * PPS], h, tid, Kstage, Vstage);

  // ---- preload Q fragments (pre-scaled, bf16, A-matrix layout) ----
  uint32_t qf[4][8];
  {
    const int row = s * QLEN + m0 + ln;
    const float* qrow = q + (((size_t)row * NHEAD + h) * NGRP + g) * DHEAD;
#pragma unroll
    for (int c = 0; c < 4; ++c)
#pragma unroll
      for (int v = 0; v < 8; ++v) {
        int d = 32 * c + (v < 4 ? 2 * v : 2 * v + 8) + 8 * half;
        float2 t = *reinterpret_cast<const float2*>(qrow + d);
        qf[c][v] = pack2(t.x * sm, t.y * sm);
      }
  }

  v8f zf = {};
  v8f acc[8];
#pragma unroll
  for (int dt = 0; dt < 8; ++dt) acc[dt] = zf;
  float mrow[8], lrow[8];
#pragma unroll
  for (int r = 0; r < 8; ++r) { mrow[r] = -1e30f; lrow[r] = 0.f; }

  const int wbase = wave * 16 * 64;

  for (int kc = 0; kc < nchunks; ++kc) {
    const int kbase = kc << 6;

    // ---- chunk kc staged: wait DMA, convert f32 stage -> bf16 buffers ----
    wait_async();
    __syncthreads();
#pragma unroll
    for (int i = 0; i < 8; ++i) {              // K: pack pairs along d
      int slot = tid + (i << 8);
      int n = slot >> 5, dq = (slot & 31) << 2;
      float4 kd = *reinterpret_cast<const float4*>(&Kstage[n * 128 + dq]);
      uint2 kp; kp.x = pack2(kd.x, kd.y); kp.y = pack2(kd.z, kd.w);
      *reinterpret_cast<uint2*>(&Kl[n * 64 + (dq >> 1)]) = kp;
    }
#pragma unroll
    for (int i = 0; i < 2; ++i) {              // V: 4x4 register transpose
      int slot = tid + (i << 8);               // 512 micro-tiles
      int d0 = (slot >> 4) << 2, n0 = (slot & 15) << 2;
      float rr[4][4];
#pragma unroll
      for (int j = 0; j < 4; ++j) {
        float4 t = *reinterpret_cast<const float4*>(&Vstage[(n0 + j) * 128 + d0]);
        rr[j][0] = t.x; rr[j][1] = t.y; rr[j][2] = t.z; rr[j][3] = t.w;
      }
#pragma unroll
      for (int dd = 0; dd < 4; ++dd) {
        uint2 vp; vp.x = pack2(rr[0][dd], rr[1][dd]);
                  vp.y = pack2(rr[2][dd], rr[3][dd]);
        *reinterpret_cast<uint2*>(&Vt[(d0 + dd) * 64 + n0]) = vp;
      }
    }
    __syncthreads();

    // ---- prefetch chunk kc+1 into stage (overlaps with WMMA below) ----
    if (kc + 1 < nchunks)
      issue_chunk_load(kv, page_idx[s * PPS + kc + 1], h, tid, Kstage, Vstage);

    // ---- scores: S = Q * K^T, 4 tiles of 16x16, k-depth 128 = 4 x 32 ----
    v8f sc[4];
#pragma unroll
    for (int t = 0; t < 4; ++t) {
      v8f c = zf;
      const int n = 16 * t + ln;
#pragma unroll
      for (int cc = 0; cc < 4; ++cc) {
        Frag a, b;
#pragma unroll
        for (int v = 0; v < 8; ++v) {
          a.u[v] = qf[cc][v];
          b.u[v] = Kl[n * 64 + 16 * cc + (v < 4 ? v : v + 4) + 4 * half];
        }
        c = __builtin_amdgcn_wmma_f32_16x16x32_bf16(false, a.v, false, b.v,
                                                    (short)0, c, false, false);
      }
      sc[t] = c;
    }

    // ---- online softmax (row = r + 8*half in C/D layout) ----
#pragma unroll
    for (int r = 0; r < 8; ++r) {
      const int qpos = kvlen - QLEN + m0 + r + 8 * half;
      float mx = -1e30f;
#pragma unroll
      for (int t = 0; t < 4; ++t) {             // mask in place
        int n = kbase + 16 * t + ln;
        float x = (n <= qpos) ? sc[t][r] : -1e30f;
        sc[t][r] = x;
        mx = fmaxf(mx, x);
      }
      mx = fmaxf(mx, __shfl_xor(mx, 1, 32));
      mx = fmaxf(mx, __shfl_xor(mx, 2, 32));
      mx = fmaxf(mx, __shfl_xor(mx, 4, 32));
      mx = fmaxf(mx, __shfl_xor(mx, 8, 32));
      float mn  = fmaxf(mrow[r], mx);
      float fac = __expf(mrow[r] - mn);
      mrow[r] = mn;
      float rs = 0.f;
#pragma unroll
      for (int t = 0; t < 4; ++t) {
        float p = __expf(sc[t][r] - mn);        // masked entries underflow to 0
        Pl[wbase + (r + 8 * half) * 64 + 16 * t + ln] = (unsigned short)bf16of(p);
        rs += p;
      }
      rs += __shfl_xor(rs, 1, 32);
      rs += __shfl_xor(rs, 2, 32);
      rs += __shfl_xor(rs, 4, 32);
      rs += __shfl_xor(rs, 8, 32);
      lrow[r] = lrow[r] * fac + rs;
#pragma unroll
      for (int dt = 0; dt < 8; ++dt) acc[dt][r] *= fac;
    }

    // ---- O += P * V : 8 d-tiles x 2 k-sub-chunks of 32 ----
#pragma unroll
    for (int dt = 0; dt < 8; ++dt) {
      const int n = 16 * dt + ln;
#pragma unroll
      for (int sub = 0; sub < 2; ++sub) {
        Frag a, b;
#pragma unroll
        for (int v = 0; v < 8; ++v) {
          int kd = 32 * sub + (v < 4 ? 2 * v : 2 * v + 8) + 8 * half;
          a.u[v] = *reinterpret_cast<const uint32_t*>(&Pl[wbase + ln * 64 + kd]);
          b.u[v] = *reinterpret_cast<const uint32_t*>(&Vt[n * 64 + kd]);
        }
        acc[dt] = __builtin_amdgcn_wmma_f32_16x16x32_bf16(false, a.v, false, b.v,
                                                          (short)0, acc[dt], false, false);
      }
    }
  }

  // ---- epilogue: O / l ----
#pragma unroll
  for (int r = 0; r < 8; ++r) {
    float inv = 1.f / fmaxf(lrow[r], 1e-10f);
    int row = s * QLEN + m0 + r + 8 * half;
    float* orow = out + (((size_t)row * NHEAD + h) * NGRP + g) * DHEAD;
#pragma unroll
    for (int dt = 0; dt < 8; ++dt)
      orow[16 * dt + ln] = acc[dt][r] * inv;
  }
}

extern "C" void kernel_launch(void* const* d_in, const int* in_sizes, int n_in,
                              void* d_out, int out_size, void* d_ws, size_t ws_size,
                              hipStream_t stream) {
  (void)in_sizes; (void)n_in; (void)out_size; (void)d_ws; (void)ws_size;
  const float* q        = (const float*)d_in[0];
  const float* kv_pages = (const float*)d_in[1];
  const int*   kv_lens  = (const int*)d_in[2];
  const int*   page_idx = (const int*)d_in[3];
  // d_in[4] = cu_q_lens (unused), d_in[5] = num_seqs (fixed at 16)
  float* out = (float*)d_out;
  dim3 grid(NSEQ * NHEAD * NGRP);   // 512 blocks: one (seq, head, group) each
  dim3 block(256);                  // 8 wave32s; wave w owns query rows 16w..16w+15
  ragged_paged_attn_kernel<<<grid, block, 0, stream>>>(q, kv_pages, kv_lens, page_idx, out);
}
